// CGConv_17918603558964
// MI455X (gfx1250) — matrix operand order repaired and checked by reading, not compiled, not run on hardware
//
#include <hip/hip_runtime.h>

typedef __attribute__((ext_vector_type(16))) __bf16 v16bf;
typedef __attribute__((ext_vector_type(8)))  float  v8f;

union Frag { v16bf v; uint4 q[2]; };

#define ATOM 128
#define NBR 64
#define DIN 320           // 2*ATOM + NBR
#define DOUT 256
#define A_STRIDE 328      // halves; padded (8*41, 41 odd) for conflict-free ds_load_b128
#define TILE_E 128        // edges per workgroup (4 mPairs x 32 rows)
#define KBLOCKS 10        // 320 / 32

__device__ __forceinline__ unsigned short f2bf(float f) {
    unsigned u = __float_as_uint(f);
    unsigned r = u + 0x7FFFu + ((u >> 16) & 1u);   // round-to-nearest-even
    return (unsigned short)(r >> 16);
}

// ---- stage 128 edges' concat features [xi|xj|ea] into LDS as bf16 ---------
// indices pre-staged in sIdx: [0..127]=src, [128..255]=dst
__device__ __forceinline__ void stage_tile(const float* __restrict__ x,
                                           const float* __restrict__ ea,
                                           unsigned short* sA,
                                           const long long* sIdx,
                                           long long eBase, long long E, int tid) {
    // 128 edges * 80 float4-slots = 10240 slots; 40 iters of 256 threads
    for (int it = 0; it < 40; ++it) {
        int s    = tid + it * 256;
        int e    = s / 80;
        int part = s % 80;                 // float4 index within concat row
        long long edge = eBase + e;
        float4 v = make_float4(0.f, 0.f, 0.f, 0.f);
        if (edge < E) {
            const float* p;
            if (part < 32)      p = x  + sIdx[e] * ATOM + part * 4;
            else if (part < 64) p = x  + sIdx[TILE_E + e] * ATOM + (part - 32) * 4;
            else                p = ea + edge * NBR + (part - 64) * 4;
            v = *(const float4*)p;
        }
        union { unsigned short h[4]; uint2 u; } pk;
        pk.h[0] = f2bf(v.x); pk.h[1] = f2bf(v.y);
        pk.h[2] = f2bf(v.z); pk.h[3] = f2bf(v.w);
        *(uint2*)(sA + e * A_STRIDE + part * 4) = pk.u;
    }
}

// ---- per-wave GEMM: 32 rows x 128 cols, K=320, bf16 WMMA ------------------
// Two 16-row groups share each B fragment (halves L2 B traffic).
__device__ __forceinline__ void gemm_tile(const unsigned short* sA,
                                          const unsigned short* __restrict__ wpack,
                                          int lane, int mPair, int nbBase,
                                          v8f acc[2][8]) {
    int laneHi = lane >> 4;
    const unsigned short* aRow0 = sA + (mPair * 32 + (lane & 15)) * A_STRIDE;
    const unsigned short* aRow1 = aRow0 + 16 * A_STRIDE;
    int kShift = laneHi * 8;   // ISA A layout: lane>=16 holds K+8..15 / K+24..31
    for (int kb = 0; kb < KBLOCKS; ++kb) {
        Frag a0, a1;
        const unsigned short* ap0 = aRow0 + kb * 32 + kShift;
        const unsigned short* ap1 = aRow1 + kb * 32 + kShift;
        a0.q[0] = *(const uint4*)(ap0);
        a0.q[1] = *(const uint4*)(ap0 + 16);
        a1.q[0] = *(const uint4*)(ap1);
        a1.q[1] = *(const uint4*)(ap1 + 16);
#pragma unroll
        for (int t = 0; t < 8; ++t) {
            Frag b;
            const unsigned short* bp = wpack + (((kb * 16) + (nbBase + t)) << 9) + lane * 16;
            b.q[0] = *(const uint4*)(bp);
            b.q[1] = *(const uint4*)(bp + 8);
            acc[0][t] = __builtin_amdgcn_wmma_f32_16x16x32_bf16(
                false, a0.v, false, b.v, (short)0, acc[0][t], false, false);
            acc[1][t] = __builtin_amdgcn_wmma_f32_16x16x32_bf16(
                false, a1.v, false, b.v, (short)0, acc[1][t], false, false);
        }
    }
}

// ---------------- kernel 0: zero output accumulator + stats ----------------
__global__ void __launch_bounds__(256) k_init(float* out, float* stats, long long total) {
    long long gid    = (long long)blockIdx.x * 256 + threadIdx.x;
    long long stride = (long long)gridDim.x * 256;
    for (long long i = gid; i < total; i += stride) out[i] = 0.f;
    if (gid < 768) stats[gid] = 0.f;   // sum1(256) sq1(256) sum2(128) sq2(128)
}

// ---------------- kernel 0b: pack W (f32 row-major) into WMMA-B bf16 -------
__global__ void __launch_bounds__(256) k_pack(const float* __restrict__ W,
                                              unsigned short* __restrict__ wpack) {
    int o = blockIdx.x * 256 + threadIdx.x;        // 0 .. 81919
    int blk = o >> 9, within = o & 511;
    int kb = blk >> 4, nb = blk & 15;
    int l = within >> 4, h = within & 15;
    int k = kb * 32 + (l >> 4) * 8 + (h & 7) + (h >> 3) * 16;   // ISA B layout: lanes>=16 hold K+8..15 / K+24..31
    int n = nb * 16 + (l & 15);
    wpack[o] = f2bf(W[k * DOUT + n]);
}

// ---------------- kernel 1: GEMM + BN1 batch statistics --------------------
__global__ void __launch_bounds__(256) k_gemm_stats(
        const float* __restrict__ x, const long long* __restrict__ ei,
        const float* __restrict__ ea, const unsigned short* __restrict__ wpack,
        const float* __restrict__ bias, float* __restrict__ gsum,
        float* __restrict__ gsq, long long E) {
    __shared__ __align__(16) unsigned short sA[TILE_E * A_STRIDE];  // 83968 B
    __shared__ float sSum[DOUT], sSq[DOUT];
    __shared__ long long sIdx[2 * TILE_E];
    int tid = threadIdx.x;
    long long eBase = (long long)blockIdx.x * TILE_E;
    sSum[tid] = 0.f; sSq[tid] = 0.f;
    {   // stage edge indices once: [0..127]=src, [128..255]=dst
        long long edge = eBase + (tid & (TILE_E - 1));
        const long long* basep = (tid < TILE_E) ? ei : (ei + E);
        sIdx[tid] = (edge < E) ? basep[edge] : 0;
    }
    __syncthreads();
    stage_tile(x, ea, sA, sIdx, eBase, E, tid);
    __syncthreads();

    int wave = tid >> 5, lane = tid & 31;
    int mPair = wave >> 1, nbBase = (wave & 1) * 8;
    int laneHi = lane >> 4;
    v8f acc[2][8];
#pragma unroll
    for (int g = 0; g < 2; ++g)
#pragma unroll
        for (int t = 0; t < 8; ++t)
#pragma unroll
            for (int r = 0; r < 8; ++r) acc[g][t][r] = 0.f;

    gemm_tile(sA, wpack, lane, mPair, nbBase, acc);

#pragma unroll
    for (int g = 0; g < 2; ++g)
#pragma unroll
        for (int t = 0; t < 8; ++t) {
            int col = (nbBase + t) * 16 + (lane & 15);
            float bv = bias[col];
            float s = 0.f, q = 0.f;
#pragma unroll
            for (int r = 0; r < 8; ++r) {
                long long edge = eBase + mPair * 32 + g * 16 + r + laneHi * 8;
                if (edge < E) { float v = acc[g][t][r] + bv; s += v; q += v * v; }
            }
            atomicAdd(&sSum[col], s);
            atomicAdd(&sSq[col], q);
        }
    __syncthreads();
    atomicAdd(&gsum[tid], sSum[tid]);
    atomicAdd(&gsq[tid],  sSq[tid]);
}

// ---------------- kernel 2: finalize BN1 -> fused scale/shift (bias folded)
__global__ void k_finalize1(const float* gsum, const float* gsq, const float* bias,
                            const float* gamma, const float* beta,
                            float* scale1, float* shiftF, float Ef) {
    int t = threadIdx.x;
    float m   = gsum[t] / Ef;
    float var = gsq[t] / Ef - m * m;
    float sc  = rsqrtf(var + 1e-5f) * gamma[t];
    scale1[t] = sc;
    shiftF[t] = beta[t] + (bias[t] - m) * sc;   // gn = acc*sc + shiftF
}

// ---------------- kernel 3: GEMM + BN1 + sigmoid*softplus + scatter-sum ----
__global__ void __launch_bounds__(256) k_gemm_scatter(
        const float* __restrict__ x, const long long* __restrict__ ei,
        const float* __restrict__ ea, const unsigned short* __restrict__ wpack,
        const float* __restrict__ scale1, const float* __restrict__ shiftF,
        float* __restrict__ out, long long E) {
    __shared__ __align__(16) unsigned short smem[TILE_E * A_STRIDE]; // 83968B; reused as sp (65536B)
    __shared__ long long sIdx[2 * TILE_E];
    int tid = threadIdx.x;
    long long eBase = (long long)blockIdx.x * TILE_E;
    {
        long long edge = eBase + (tid & (TILE_E - 1));
        const long long* basep = (tid < TILE_E) ? ei : (ei + E);
        sIdx[tid] = (edge < E) ? basep[edge] : 0;
    }
    __syncthreads();
    stage_tile(x, ea, smem, sIdx, eBase, E, tid);
    __syncthreads();

    int wave = tid >> 5, lane = tid & 31;
    int mPair = wave >> 1, half = wave & 1;   // half 0: filter cols 0..127, half 1: core cols 128..255
    int nbBase = half * 8;
    int laneHi = lane >> 4;
    v8f acc[2][8];
#pragma unroll
    for (int g = 0; g < 2; ++g)
#pragma unroll
        for (int t = 0; t < 8; ++t)
#pragma unroll
            for (int r = 0; r < 8; ++r) acc[g][t][r] = 0.f;

    gemm_tile(smem, wpack, lane, mPair, nbBase, acc);
    __syncthreads();                              // done reading A tile; reuse LDS
    float* sp = (float*)smem;                     // softplus buffer [128][128]

    if (half == 1) {                              // core half -> softplus to LDS
#pragma unroll
        for (int t = 0; t < 8; ++t) {
            int colc = t * 16 + (lane & 15);
            float sc = scale1[128 + colc], sh = shiftF[128 + colc];
#pragma unroll
            for (int g = 0; g < 2; ++g)
#pragma unroll
                for (int r = 0; r < 8; ++r) {
                    int rowL = mPair * 32 + g * 16 + r + laneHi * 8;
                    float gn = acc[g][t][r] * sc + sh;
                    sp[rowL * ATOM + colc] = fmaxf(gn, 0.f) + log1pf(__expf(-fabsf(gn)));
                }
        }
    }
    __syncthreads();
    if (half == 0) {                              // filter half -> sigmoid * sp, scatter
        float scv[8], shv[8];
#pragma unroll
        for (int t = 0; t < 8; ++t) {
            int col = t * 16 + (lane & 15);
            scv[t] = scale1[col]; shv[t] = shiftF[col];
        }
#pragma unroll
        for (int g = 0; g < 2; ++g)
#pragma unroll
            for (int r = 0; r < 8; ++r) {
                int rowL = mPair * 32 + g * 16 + r + laneHi * 8;
                long long edge = eBase + rowL;
                if (edge < E) {
                    long long node = sIdx[rowL];  // segment index = src
                    float* orow = out + node * ATOM;
#pragma unroll
                    for (int t = 0; t < 8; ++t) {
                        int col = t * 16 + (lane & 15);
                        float gn  = acc[g][t][r] * scv[t] + shv[t];
                        float sg  = 1.f / (1.f + __expf(-gn));
                        atomicAdd(orow + col, sg * sp[rowL * ATOM + col]);
                    }
                }
            }
    }
}

// ---------------- kernel 4: BN2 column stats over summed [N,128] -----------
__global__ void __launch_bounds__(256) k_stats2(const float* __restrict__ out,
                                                float* gsum2, float* gsq2, long long N) {
    __shared__ float s2[ATOM], q2[ATOM];
    int tid = threadIdx.x;
    if (tid < ATOM) { s2[tid] = 0.f; q2[tid] = 0.f; }
    __syncthreads();
    int col = tid & (ATOM - 1);
    float s = 0.f, q = 0.f;
    for (long long r = (long long)blockIdx.x * 2 + (tid >> 7); r < N;
         r += (long long)gridDim.x * 2) {
        float v = out[r * ATOM + col];
        s += v; q += v * v;
    }
    atomicAdd(&s2[col], s);
    atomicAdd(&q2[col], q);
    __syncthreads();
    if (tid < ATOM) { atomicAdd(&gsum2[tid], s2[tid]); atomicAdd(&gsq2[tid], q2[tid]); }
}

// ---------------- kernel 5: BN2 normalize in place -------------------------
__global__ void __launch_bounds__(256) k_bn2(float* out, const float* gsum2,
                                             const float* gsq2, const float* g2,
                                             const float* be2, long long N) {
    long long idx = (long long)blockIdx.x * 256 + threadIdx.x;
    if (idx < N * ATOM) {
        int col = (int)(idx & (ATOM - 1));
        float m   = gsum2[col] / (float)N;
        float var = gsq2[col] / (float)N - m * m;
        float inv = rsqrtf(var + 1e-5f);
        out[idx] = (out[idx] - m) * inv * g2[col] + be2[col];
    }
}

extern "C" void kernel_launch(void* const* d_in, const int* in_sizes, int n_in,
                              void* d_out, int out_size, void* d_ws, size_t ws_size,
                              hipStream_t stream) {
    const float*     x   = (const float*)d_in[0];
    const long long* ei  = (const long long*)d_in[1];   // [2,E] int64: src row then dst row
    const float*     ea  = (const float*)d_in[2];
    const float*     W   = (const float*)d_in[3];
    const float*     b   = (const float*)d_in[4];
    const float*     g1  = (const float*)d_in[5];
    const float*     be1 = (const float*)d_in[6];
    const float*     g2  = (const float*)d_in[7];
    const float*     be2 = (const float*)d_in[8];
    float* out = (float*)d_out;

    long long N = in_sizes[0] / ATOM;
    long long E = in_sizes[1] / 2;

    char* ws = (char*)d_ws;
    unsigned short* wpack = (unsigned short*)ws;              // 320*256*2 = 163840 B
    float* stats  = (float*)(ws + DIN * DOUT * 2);
    float* gsum1  = stats;            // 256
    float* gsq1   = stats + 256;      // 256
    float* gsum2  = stats + 512;      // 128
    float* gsq2   = stats + 640;      // 128
    float* scale1 = stats + 768;      // 256
    float* shiftF = stats + 1024;     // 256

    long long total = N * ATOM;
    int eBlocks = (int)((E + TILE_E - 1) / TILE_E);

    k_init<<<2048, 256, 0, stream>>>(out, stats, total);
    k_pack<<<(DIN * DOUT) / 256, 256, 0, stream>>>(W, wpack);
    k_gemm_stats<<<eBlocks, 256, 0, stream>>>(x, ei, ea, wpack, b, gsum1, gsq1, E);
    k_finalize1<<<1, DOUT, 0, stream>>>(gsum1, gsq1, b, g1, be1, scale1, shiftF, (float)E);
    k_gemm_scatter<<<eBlocks, 256, 0, stream>>>(x, ei, ea, wpack, scale1, shiftF, out, E);
    k_stats2<<<256, 256, 0, stream>>>(out, gsum2, gsq2, N);
    k_bn2<<<(int)((total + 255) / 256), 256, 0, stream>>>(out, gsum2, gsq2, g2, be2, N);
}
